// BERT_BiLSTM_CRF_37383395344938
// MI455X (gfx1250) — compile-verified, matrix operand order checked by reading, not run on hardware
//
#include <hip/hip_runtime.h>
#include <hip/hip_bf16.h>

// ---------------------------------------------------------------------------
// BERT-BiLSTM-CRF for MI455X (gfx1250, wave32, WMMA, async LDS staging).
// B=64, T=512, D=768, H=384, K=16.  All GEMMs in bf16 WMMA (f32 accum).
// ---------------------------------------------------------------------------

typedef __bf16 bf16_t;
typedef __attribute__((ext_vector_type(8)))  bf16_t v8bf;
typedef __attribute__((ext_vector_type(16))) bf16_t v16bf;
typedef __attribute__((ext_vector_type(8)))  float  v8f;
typedef __attribute__((ext_vector_type(4)))  int    v4i;

#if defined(__AMDGCN__)
typedef __attribute__((address_space(1))) v4i* gas_v4i_ptr;  // global
typedef __attribute__((address_space(3))) v4i* las_v4i_ptr;  // LDS
#endif

#define T_SEQ 512
#define DIM   768
#define HID   384
#define GATES 1536
#define NTAGS 16
#define BSZ   64
#define M_ROWS (BSZ * T_SEQ)   // 32768
#define START_TAG 14
#define NEG_INF  -10000.0f

#if defined(__gfx1250__) && __has_builtin(__builtin_amdgcn_global_load_async_to_lds_b128)
#define HAS_ASYNC_LDS 1
#else
#define HAS_ASYNC_LDS 0
#endif

// --- WMMA helpers -----------------------------------------------------------

__device__ __forceinline__ v8f wmma_bf16(v16bf a, v16bf b, v8f c) {
    // D = A(16x32) * B(32x16) + C(16x16), f32 accumulate
    return __builtin_amdgcn_wmma_f32_16x16x32_bf16(
        /*neg_a=*/false, a, /*neg_b=*/false, b,
        /*c_mod=*/(short)0, c, /*reuse_a=*/false, /*reuse_b=*/false);
}

// Load a 16x32 bf16 operand fragment in the ISA-specified per-lane layout:
// lanes 0-15 hold row (lane&15), K chunks {k0..k0+7, k0+16..k0+23};
// lanes 16-31 hold the same rows,  K chunks {k0+8..k0+15, k0+24..k0+31}.
__device__ __forceinline__ v16bf load_frag(const bf16_t* __restrict__ base,
                                           int stride, int row, int k0, int khalf) {
    const bf16_t* p = base + (size_t)row * stride + k0 + khalf;
    v8bf lo = *(const v8bf*)(p);
    v8bf hi = *(const v8bf*)(p + 16);
    return __builtin_shufflevector(lo, hi, 0,1,2,3,4,5,6,7,8,9,10,11,12,13,14,15);
}

// 16-byte global -> LDS transfer: async (ASYNCcnt) on gfx1250, sync fallback.
__device__ __forceinline__ void stage16(const bf16_t* gsrc, bf16_t* lds_dst) {
#if HAS_ASYNC_LDS
    __builtin_amdgcn_global_load_async_to_lds_b128(
        (gas_v4i_ptr)gsrc, (las_v4i_ptr)lds_dst, /*offset=*/0, /*cpol=*/0);
#else
    *(v8bf*)lds_dst = *(const v8bf*)gsrc;
#endif
}

template <int N>
__device__ __forceinline__ void wait_async() {
#if HAS_ASYNC_LDS
#if __has_builtin(__builtin_amdgcn_s_wait_asynccnt)
    __builtin_amdgcn_s_wait_asynccnt((short)N);
#else
    asm volatile("s_wait_asynccnt %0" :: "i"(N) : "memory");
#endif
#endif
}

__device__ __forceinline__ float sigm(float x) { return 1.0f / (1.0f + __expf(-x)); }

// --- f32 -> bf16 conversion -------------------------------------------------

__global__ void cvt_bf16_kernel(const float* __restrict__ src,
                                bf16_t* __restrict__ dst, int n) {
    int i = blockIdx.x * blockDim.x + threadIdx.x;
    int stride = gridDim.x * blockDim.x;
    for (; i < n; i += stride) dst[i] = (bf16_t)src[i];
}

// --- Input-side gate GEMM ---------------------------------------------------
// xg[M=32768, N=1536] = X[M, 768] * Wih^T + (b_ih + b_hh)
// Block tile 128x128, BK=32, 256 threads = 8 waves (2M x 4N), each wave a
// 64x32 sub-tile (4x2 WMMA accumulators).  K-slices double-buffered in LDS
// via GLOBAL_LOAD_ASYNC_TO_LDS_B128 (4 x 16B per thread per slice).
// __launch_bounds__(256, 2): keep accumulators register-resident (no spills);
// 2 waves/SIMD is enough to hide WMMA latency with 8 independent WMMAs/wave.

__global__ void __launch_bounds__(256, 2)
gate_gemm_kernel(const bf16_t* __restrict__ X,
                 const bf16_t* __restrict__ Wih,
                 const float*  __restrict__ bih,
                 const float*  __restrict__ bhh,
                 float*        __restrict__ xg) {
    __shared__ bf16_t As[2 * 128 * 32];   // [buf][m][k]
    __shared__ bf16_t Bs[2 * 128 * 32];   // [buf][n][k]

    const int tid   = threadIdx.x;
    const int w     = tid >> 5;
    const int lane  = tid & 31;
    const int ln    = lane & 15;
    const int khalf = (lane >> 4) * 8;
    const int hi8   = khalf;           // C-layout row offset for lanes 16-31
    const int m0    = blockIdx.y * 128;
    const int n0    = blockIdx.x * 128;
    const int wm    = w >> 2;          // 0..1
    const int wn    = w & 3;           // 0..3

    v8f acc[4][2];
    #pragma unroll
    for (int mt = 0; mt < 4; ++mt)
        #pragma unroll
        for (int nt = 0; nt < 2; ++nt)
            #pragma unroll
            for (int r = 0; r < 8; ++r) acc[mt][nt][r] = 0.0f;

    // Per-thread staging: 2 chunks of A + 2 chunks of B (16B each) per slice.
    const int c0   = tid * 2;               // chunk ids c0, c0+1 (0..511)
    const int row0 = c0 >> 2,       off0 = (c0 & 3) * 8;
    const int row1 = (c0 + 1) >> 2, off1 = ((c0 + 1) & 3) * 8;

    auto stage_slice = [&](int k0, int buf) {
        bf16_t* a = As + buf * (128 * 32);
        bf16_t* b = Bs + buf * (128 * 32);
        stage16(X   + (size_t)(m0 + row0) * DIM + k0 + off0, a + row0 * 32 + off0);
        stage16(Wih + (size_t)(n0 + row0) * DIM + k0 + off0, b + row0 * 32 + off0);
        stage16(X   + (size_t)(m0 + row1) * DIM + k0 + off1, a + row1 * 32 + off1);
        stage16(Wih + (size_t)(n0 + row1) * DIM + k0 + off1, b + row1 * 32 + off1);
    };

    auto compute_slice = [&](int buf) {
        const bf16_t* a = As + buf * (128 * 32);
        const bf16_t* b = Bs + buf * (128 * 32);
        v16bf av[4], bv[2];
        #pragma unroll
        for (int mt = 0; mt < 4; ++mt)
            av[mt] = load_frag(a, 32, wm * 64 + mt * 16 + ln, 0, khalf);
        #pragma unroll
        for (int nt = 0; nt < 2; ++nt)
            bv[nt] = load_frag(b, 32, wn * 32 + nt * 16 + ln, 0, khalf);
        #pragma unroll
        for (int mt = 0; mt < 4; ++mt)
            #pragma unroll
            for (int nt = 0; nt < 2; ++nt)
                acc[mt][nt] = wmma_bf16(av[mt], bv[nt], acc[mt][nt]);
    };

    constexpr int KSTEPS = DIM / 32;   // 24
    stage_slice(0, 0);
    for (int it = 0; it < KSTEPS - 1; ++it) {
        stage_slice((it + 1) * 32, (it + 1) & 1);  // prefetch next slice
        wait_async<4>();                           // drain current slice's 4 loads
        __syncthreads();
        compute_slice(it & 1);
        __syncthreads();                           // reads done before buf reuse
    }
    wait_async<0>();
    __syncthreads();
    compute_slice((KSTEPS - 1) & 1);

    // Epilogue: add bias, store f32 gates.
    #pragma unroll
    for (int nt = 0; nt < 2; ++nt) {
        const int n = n0 + wn * 32 + nt * 16 + ln;
        const float bias = bih[n] + bhh[n];
        #pragma unroll
        for (int mt = 0; mt < 4; ++mt) {
            #pragma unroll
            for (int r = 0; r < 8; ++r) {
                const int m = m0 + wm * 64 + mt * 16 + r + hi8;
                xg[(size_t)m * GATES + n] = acc[mt][nt][r] + bias;
            }
        }
    }
}

// --- LSTM recurrence --------------------------------------------------------
// One block = 16 batch rows (4 blocks per direction).  Per step:
// G[16,1536] = xg_t + h[16,384] @ Whh^T, then gate nonlinearity.
// h kept in LDS (bf16, WMMA A operand shared by all waves); c in VGPRs.
// Each wave owns hidden slice [w*48, w*48+48) across ALL four gates so the
// i/f/g/o coupling stays register-resident.  Whh (1.18 MB bf16) streams from
// L2 every step.  __launch_bounds__(256, 2): acc[4][3]+creg[3] = 120 VGPRs of
// persistent state must not spill.

__global__ void __launch_bounds__(256, 2)
lstm_rec_kernel(const float*  __restrict__ xg,
                const bf16_t* __restrict__ whh,
                bf16_t*       __restrict__ hcat,
                int dir) {
    __shared__ bf16_t hbuf[16 * HID];

    const int tid   = threadIdx.x;
    const int w     = tid >> 5;
    const int lane  = tid & 31;
    const int ln    = lane & 15;
    const int khalf = (lane >> 4) * 8;
    const int hi8   = khalf;
    const int b0    = blockIdx.x * 16;
    const int jbase = w * 48;

    for (int i = tid; i < 16 * HID; i += blockDim.x) hbuf[i] = (bf16_t)0.0f;

    v8f creg[3];
    #pragma unroll
    for (int tt = 0; tt < 3; ++tt)
        #pragma unroll
        for (int r = 0; r < 8; ++r) creg[tt][r] = 0.0f;

    __syncthreads();

    for (int s = 0; s < T_SEQ; ++s) {
        const int t = dir ? (T_SEQ - 1 - s) : s;

        // Accumulators init = precomputed input gates xg_t (f32, C layout).
        v8f acc[4][3];
        #pragma unroll
        for (int g = 0; g < 4; ++g) {
            #pragma unroll
            for (int tt = 0; tt < 3; ++tt) {
                const int n = g * HID + jbase + tt * 16 + ln;
                #pragma unroll
                for (int r = 0; r < 8; ++r) {
                    const int m = r + hi8;
                    acc[g][tt][r] =
                        xg[((size_t)(b0 + m) * T_SEQ + t) * GATES + n];
                }
            }
        }

        // += h @ Whh^T : K=384 in 12 steps; A fragment reused across 12 N-tiles.
        for (int kk = 0; kk < HID; kk += 32) {
            v16bf av = load_frag(hbuf, HID, ln, kk, khalf);
            #pragma unroll
            for (int g = 0; g < 4; ++g) {
                #pragma unroll
                for (int tt = 0; tt < 3; ++tt) {
                    const int n = g * HID + jbase + tt * 16 + ln;
                    v16bf bv = load_frag(whh, HID, n, kk, khalf);
                    acc[g][tt] = wmma_bf16(av, bv, acc[g][tt]);
                }
            }
        }

        __syncthreads();   // all waves finished reading hbuf(t-1)

        // Gate nonlinearity, cell update, write h (LDS bf16 + global bf16).
        #pragma unroll
        for (int tt = 0; tt < 3; ++tt) {
            const int j = jbase + tt * 16 + ln;
            #pragma unroll
            for (int r = 0; r < 8; ++r) {
                const float ig = sigm(acc[0][tt][r]);
                const float fg = sigm(acc[1][tt][r]);
                const float gg = tanhf(acc[2][tt][r]);
                const float og = sigm(acc[3][tt][r]);
                const float c  = fg * creg[tt][r] + ig * gg;
                const float h  = og * tanhf(c);
                creg[tt][r] = c;
                const int m = r + hi8;
                hbuf[m * HID + j] = (bf16_t)h;
                hcat[((size_t)(b0 + m) * T_SEQ + t) * (2 * HID) + dir * HID + j]
                    = (bf16_t)h;
            }
        }
        __syncthreads();   // hbuf(t) ready for next step
    }
}

// --- Emission projection ----------------------------------------------------
// feats[M=32768, 16] = hcat[M, 768] @ w_out^T + b_out   (one wave per M tile)

__global__ void feats_kernel(const bf16_t* __restrict__ hcat,
                             const bf16_t* __restrict__ wout,
                             const float*  __restrict__ bout,
                             float*        __restrict__ feats) {
    const int tid   = threadIdx.x;
    const int w     = tid >> 5;
    const int lane  = tid & 31;
    const int ln    = lane & 15;
    const int khalf = (lane >> 4) * 8;
    const int hi8   = khalf;
    const int m0    = (blockIdx.x * 8 + w) * 16;

    v8f acc;
    #pragma unroll
    for (int r = 0; r < 8; ++r) acc[r] = 0.0f;

    for (int k0 = 0; k0 < DIM; k0 += 32) {
        v16bf av = load_frag(hcat, DIM, m0 + ln, k0, khalf);
        v16bf bv = load_frag(wout, DIM, ln,      k0, khalf);
        acc = wmma_bf16(av, bv, acc);
    }
    const float bias = bout[ln];
    #pragma unroll
    for (int r = 0; r < 8; ++r)
        feats[(size_t)(m0 + r + hi8) * NTAGS + ln] = acc[r] + bias;
}

// --- Viterbi ----------------------------------------------------------------
// One wave32 per batch row.  delta[k] lives in lane k's register; the
// k_prev sweep uses __shfl (ds_bpermute).  Backtrace serial on lane 0.

__global__ void viterbi_kernel(const float* __restrict__ feats,
                               const float* __restrict__ trans,
                               int*         __restrict__ psi,
                               float*       __restrict__ out) {
    const int b    = blockIdx.x;
    const int lane = threadIdx.x;
    const int k    = lane & 15;

    float tr[16];
    #pragma unroll
    for (int kp = 0; kp < 16; ++kp) tr[kp] = trans[k * 16 + kp];

    float delta = (k == START_TAG) ? 0.0f : NEG_INF;

    for (int t = 1; t < T_SEQ; ++t) {
        float best = -3.4e38f;
        int   arg  = 0;
        #pragma unroll
        for (int kp = 0; kp < 16; ++kp) {
            const float dp = __shfl(delta, kp, 32);
            const float sc = tr[kp] + dp;
            if (sc > best) { best = sc; arg = kp; }   // first max wins (argmax)
        }
        if (lane < 16) psi[((size_t)b * T_SEQ + t) * NTAGS + k] = arg;
        delta = best + feats[((size_t)b * T_SEQ + t) * NTAGS + k];
    }

    // argmax over the 16 tags (lanes 16-31 hold duplicates; tie rule fixes it)
    float bd = delta;
    int   bk = k;
    #pragma unroll
    for (int off = 16; off > 0; off >>= 1) {
        const float od = __shfl_down(bd, off, 32);
        const int   ok = __shfl_down(bk, off, 32);
        if (od > bd || (od == bd && ok < bk)) { bd = od; bk = ok; }
    }

    if (lane == 0) {
        out[b] = bd;
        float* path = out + BSZ + (size_t)b * T_SEQ;
        int cur = bk;
        path[T_SEQ - 1] = (float)cur;
        for (int t = T_SEQ - 2; t >= 0; --t) {
            cur = psi[((size_t)b * T_SEQ + (t + 1)) * NTAGS + cur];
            path[t] = (float)cur;
        }
    }
}

// --- Host-side orchestration ------------------------------------------------

extern "C" void kernel_launch(void* const* d_in, const int* in_sizes, int n_in,
                              void* d_out, int out_size, void* d_ws, size_t ws_size,
                              hipStream_t stream) {
    const float* encoded = (const float*)d_in[0];   // [64,512,768]
    const float* w_ih    = (const float*)d_in[1];   // [2,2,1536,768]
    const float* w_hh    = (const float*)d_in[2];   // [2,2,1536,384]
    const float* b_ih    = (const float*)d_in[3];   // [2,2,1536]
    const float* b_hh    = (const float*)d_in[4];   // [2,2,1536]
    const float* w_out   = (const float*)d_in[5];   // [16,768]
    const float* b_out   = (const float*)d_in[6];   // [16]
    const float* trans   = (const float*)d_in[7];   // [16,16]
    float* out = (float*)d_out;                     // [64] score + [64,512] path

    // Workspace layout (256B aligned partitions)
    char* ws = (char*)d_ws;
    const size_t SZ_XBF  = (size_t)M_ROWS * DIM * sizeof(bf16_t);        // 50.3 MB
    const size_t SZ_WIH  = (size_t)4 * GATES * DIM * sizeof(bf16_t);     //  9.4 MB
    const size_t SZ_WHH  = (size_t)4 * GATES * HID * sizeof(bf16_t);     //  4.7 MB
    const size_t SZ_WOUT = (size_t)NTAGS * DIM * sizeof(bf16_t);
    const size_t SZ_XG   = (size_t)M_ROWS * GATES * sizeof(float);       // 201 MB
    const size_t SZ_FTS  = (size_t)M_ROWS * NTAGS * sizeof(float);       //   2 MB

    size_t off = 0;
    auto carve = [&](size_t bytes) {
        void* p = ws + off;
        off += (bytes + 255) & ~(size_t)255;
        return p;
    };
    bf16_t* xbf0  = (bf16_t*)carve(SZ_XBF);
    bf16_t* xbf1  = (bf16_t*)carve(SZ_XBF);
    bf16_t* wihb  = (bf16_t*)carve(SZ_WIH);
    bf16_t* whhb  = (bf16_t*)carve(SZ_WHH);
    bf16_t* woutb = (bf16_t*)carve(SZ_WOUT);
    float*  xg    = (float*)carve(SZ_XG);
    float*  fts   = (float*)carve(SZ_FTS);
    int*    psi   = (int*)carve((size_t)BSZ * T_SEQ * NTAGS * sizeof(int));

    // bf16 conversions
    cvt_bf16_kernel<<<2048, 256, 0, stream>>>(encoded, xbf0, M_ROWS * DIM);
    cvt_bf16_kernel<<<1024, 256, 0, stream>>>(w_ih, wihb, 4 * GATES * DIM);
    cvt_bf16_kernel<<<512,  256, 0, stream>>>(w_hh, whhb, 4 * GATES * HID);
    cvt_bf16_kernel<<<8,    256, 0, stream>>>(w_out, woutb, NTAGS * DIM);

    // 2-layer BiLSTM (directions sequential, sharing one xg buffer)
    const bf16_t* Xin  = xbf0;
    bf16_t*       Hout = xbf1;
    for (int layer = 0; layer < 2; ++layer) {
        for (int d = 0; d < 2; ++d) {
            const int idx = layer * 2 + d;
            gate_gemm_kernel<<<dim3(GATES / 128, M_ROWS / 128), 256, 0, stream>>>(
                Xin, wihb + (size_t)idx * GATES * DIM,
                b_ih + (size_t)idx * GATES, b_hh + (size_t)idx * GATES, xg);
            lstm_rec_kernel<<<4, 256, 0, stream>>>(
                xg, whhb + (size_t)idx * GATES * HID, Hout, d);
        }
        const bf16_t* tmp = Hout;
        Hout = (bf16_t*)Xin;
        Xin  = tmp;
    }
    // After the swap, Xin points at the final layer's [B,T,768] bf16 output.

    feats_kernel<<<M_ROWS / 128, 256, 0, stream>>>(Xin, woutb, b_out, fts);
    viterbi_kernel<<<BSZ, 32, 0, stream>>>(fts, trans, psi, out);
}
